// FMTabularProcessor_88888643158051
// MI455X (gfx1250) — matrix operand order repaired and checked by reading, not compiled, not run on hardware
//
#include <hip/hip_runtime.h>
#include <hip/hip_bf16.h>

// MI455X / gfx1250, wave32. All GEMMs via v_wmma_f32_16x16x32_bf16.
typedef __bf16 bf16;
typedef __attribute__((ext_vector_type(16))) __bf16 v16bf;
typedef __attribute__((ext_vector_type(8)))  float  v8f;

// ---------------- bf16 helpers (RNE) ----------------
static __device__ __forceinline__ float bfbits2f(unsigned short h) {
  unsigned u = ((unsigned)h) << 16;
  return __builtin_bit_cast(float, u);
}
static __device__ __forceinline__ unsigned short f2bfbits(float f) {
  unsigned u = __builtin_bit_cast(unsigned, f);
  unsigned r = u + 0x7FFFu + ((u >> 16) & 1u);
  return (unsigned short)(r >> 16);
}
static __device__ __forceinline__ bf16 f2bf(float f) {
  unsigned short s = f2bfbits(f);
  return __builtin_bit_cast(bf16, s);
}
static __device__ __forceinline__ unsigned sq_pk(unsigned p) {
  float lo = bfbits2f((unsigned short)(p & 0xFFFFu));
  float hi = bfbits2f((unsigned short)(p >> 16));
  return (unsigned)f2bfbits(lo * lo) | ((unsigned)f2bfbits(hi * hi) << 16);
}
static __device__ __forceinline__ uint4 sq_q(uint4 q) {
  return make_uint4(sq_pk(q.x), sq_pk(q.y), sq_pk(q.z), sq_pk(q.w));
}

// ---------------- WMMA fragment loads (ISA 7.12.2 layouts) ----------------
// A (16-bit, 16x32): lane m=lane&15, half=lane>>4; elems 0..7 -> K=half*8+e,
// elems 8..15 -> K=16+half*8+(e-8). Row-major LDS, two b128 loads.
static __device__ __forceinline__ v16bf load_frag_a(const bf16* rowp, int half) {
  union { v16bf v; uint4 q[2]; } u;
  u.q[0] = *(const uint4*)(rowp + half * 8);
  u.q[1] = *(const uint4*)(rowp + 16 + half * 8);
  return u.v;
}
// B (16-bit, 32x16): lane n=lane&15, half=lane>>4; elems 0..7 -> K=half*8+e,
// elems 8..15 -> K=16+half*8+(e-8).
// LDS holds B transposed ([n][k], k-contiguous): two b128 loads.
static __device__ __forceinline__ v16bf load_frag_b(const bf16* rowp, int half) {
  union { v16bf v; uint4 q[2]; } u;
  u.q[0] = *(const uint4*)(rowp + half * 8);
  u.q[1] = *(const uint4*)(rowp + 16 + half * 8);
  return u.v;
}
static __device__ __forceinline__ v8f wmma_bf16(v16bf a, v16bf b, v8f c) {
  return __builtin_amdgcn_wmma_f32_16x16x32_bf16(false, a, false, b, (short)0, c,
                                                 false, false);
}
static __device__ __forceinline__ void store_out(bf16* p, float v) { *p = f2bf(v); }
static __device__ __forceinline__ void store_out(float* p, float v) { *p = v; }

// ---------------- conversion kernels ----------------
__global__ void k_cvt_bf16(const float* __restrict__ in, bf16* __restrict__ out,
                           long long n4) {
  long long i = (long long)blockIdx.x * blockDim.x + threadIdx.x;
  if (i < n4) {
    float4 f = ((const float4*)in)[i];
    unsigned lo = (unsigned)f2bfbits(f.x) | ((unsigned)f2bfbits(f.y) << 16);
    unsigned hi = (unsigned)f2bfbits(f.z) | ((unsigned)f2bfbits(f.w) << 16);
    ((uint2*)out)[i] = make_uint2(lo, hi);
  }
}
// in: [R,C] fp32 row-major -> out: [C,R] bf16 row-major (k-contiguous B^T)
__global__ void k_transpose_cvt(const float* __restrict__ in, bf16* __restrict__ out,
                                int R, int C) {
  int i = blockIdx.x * blockDim.x + threadIdx.x;
  if (i < R * C) {
    int r = i / C, c = i % C;
    out[(size_t)c * R + r] = f2bf(in[i]);
  }
}

// ---------------- generic bf16 WMMA GEMM (double-buffered LDS) ----------------
// C[M,N](+bias,relu?) = A[M,K] @ Bt[N,K]^T ; M%128==0, N%128==0, K%32==0
template <bool RELU, typename OutT>
__global__ __launch_bounds__(256) void gemm_wmma(const bf16* __restrict__ A,
                                                 const bf16* __restrict__ Bt,
                                                 const float* __restrict__ bias,
                                                 OutT* __restrict__ C, int ldc, int K) {
  constexpr int BM = 128, BN = 128, BK = 32, LDA = 40;  // 80B rows: conflict-free frags
  constexpr int ATILE = BM * LDA, BTILE = BN * LDA;
  __shared__ bf16 As[2 * ATILE];   // double buffer: one barrier per k-iter
  __shared__ bf16 Bs[2 * BTILE];

  const int tid = threadIdx.x;
  const int lane = tid & 31, lm = lane & 15, half = lane >> 4;
  const int wid = tid >> 5;
  const int wm = (wid >> 1) * 32;  // 4 waves over M
  const int wn = (wid & 1) * 64;   // 2 waves over N
  const size_t m0 = (size_t)blockIdx.x * BM;
  const size_t n0 = (size_t)blockIdx.y * BN;

  const int lrow = tid >> 1;        // 0..127
  const int lkh = (tid & 1) * 16;   // 0 / 16 elems
  const bf16* gA = A + (m0 + lrow) * (size_t)K + lkh;
  const bf16* gB = Bt + (n0 + lrow) * (size_t)K + lkh;
  const int lds_wr = lrow * LDA + lkh;

  v8f acc[2][4] = {};

  // prologue: stage tile 0 into buffer 0
  uint4 ra0 = *(const uint4*)(gA);
  uint4 ra1 = *(const uint4*)(gA + 8);
  uint4 rb0 = *(const uint4*)(gB);
  uint4 rb1 = *(const uint4*)(gB + 8);
  *(uint4*)(As + lds_wr) = ra0;
  *(uint4*)(As + lds_wr + 8) = ra1;
  *(uint4*)(Bs + lds_wr) = rb0;
  *(uint4*)(Bs + lds_wr + 8) = rb1;

  int p = 0;
  for (int k0 = 0; k0 < K; k0 += BK) {
    __syncthreads();  // buffer p ready for all waves
    const bool more = (k0 + BK) < K;
    if (more) {  // next tile's global loads fly while WMMAs run on buffer p
      ra0 = *(const uint4*)(gA + k0 + BK);
      ra1 = *(const uint4*)(gA + k0 + BK + 8);
      rb0 = *(const uint4*)(gB + k0 + BK);
      rb1 = *(const uint4*)(gB + k0 + BK + 8);
      if (k0 + 2 * BK < K) {  // WGP-near prefetch of tile k+2
        __builtin_prefetch(gA + k0 + 2 * BK, 0, 3);
        __builtin_prefetch(gB + k0 + 2 * BK, 0, 3);
      }
    }
    const bf16* as = As + p * ATILE;
    const bf16* bs = Bs + p * BTILE;

    v16bf af[2], bfr[4];
#pragma unroll
    for (int i = 0; i < 2; ++i)
      af[i] = load_frag_a(as + (wm + i * 16 + lm) * LDA, half);
#pragma unroll
    for (int j = 0; j < 4; ++j)
      bfr[j] = load_frag_b(bs + (wn + j * 16 + lm) * LDA, half);
#pragma unroll
    for (int i = 0; i < 2; ++i)
#pragma unroll
      for (int j = 0; j < 4; ++j) acc[i][j] = wmma_bf16(af[i], bfr[j], acc[i][j]);

    if (more) {
      bf16* asn = As + (p ^ 1) * ATILE;
      bf16* bsn = Bs + (p ^ 1) * BTILE;
      *(uint4*)(asn + lds_wr) = ra0;
      *(uint4*)(asn + lds_wr + 8) = ra1;
      *(uint4*)(bsn + lds_wr) = rb0;
      *(uint4*)(bsn + lds_wr + 8) = rb1;
    }
    p ^= 1;
  }

#pragma unroll
  for (int i = 0; i < 2; ++i)
#pragma unroll
    for (int j = 0; j < 4; ++j) {
      const int n = (int)n0 + wn + j * 16 + lm;
      const float bv = bias[n];
#pragma unroll
      for (int r = 0; r < 8; ++r) {  // C layout: M = r + half*8 (+ tile), N = lm
        const size_t m = m0 + wm + i * 16 + half * 8 + r;
        float v = acc[i][j][r] + bv;
        if (RELU) v = v > 0.f ? v : 0.f;
        store_out(&C[m * (size_t)ldc + n], v);
      }
    }
}

// ---------------- FM second-order kernel (double-buffered) ----------------
// combined[:, off:off+64] = 0.5*((x@V)^2 - (x^2)@(V^2)); squares built in-register.
__global__ __launch_bounds__(256) void fm_wmma(const bf16* __restrict__ X,
                                               const bf16* __restrict__ Vt,  // [64,K]
                                               bf16* __restrict__ C, int ldc,
                                               int ncol_off, int K) {
  constexpr int BM = 128, BK = 32, LDA = 40;
  constexpr int XT = BM * LDA, VT = 64 * LDA;
  __shared__ bf16 Xs[2 * XT];
  __shared__ bf16 X2s[2 * XT];
  __shared__ bf16 Vs[2 * VT];
  __shared__ bf16 V2s[2 * VT];

  const int tid = threadIdx.x;
  const int lane = tid & 31, lm = lane & 15, half = lane >> 4;
  const int wid = tid >> 5;
  const int wm = (wid >> 1) * 32;
  const int wn = (wid & 1) * 32;
  const size_t m0 = (size_t)blockIdx.x * BM;

  const int lrow = tid >> 1;
  const int lkh = (tid & 1) * 16;
  const bf16* gX = X + (m0 + lrow) * (size_t)K + lkh;
  const bf16* gV = Vt + (size_t)lrow * K + lkh;  // valid for tid<128 (64 rows)
  const int lds_wr = lrow * LDA + lkh;

  v8f acc1[2][2] = {};
  v8f acc2[2][2] = {};

  // prologue: stage tile 0 into buffer 0
  uint4 x0 = *(const uint4*)(gX);
  uint4 x1 = *(const uint4*)(gX + 8);
  uint4 v0 = make_uint4(0, 0, 0, 0), v1 = make_uint4(0, 0, 0, 0);
  if (tid < 128) {
    v0 = *(const uint4*)(gV);
    v1 = *(const uint4*)(gV + 8);
  }
  *(uint4*)(Xs + lds_wr) = x0;
  *(uint4*)(Xs + lds_wr + 8) = x1;
  *(uint4*)(X2s + lds_wr) = sq_q(x0);
  *(uint4*)(X2s + lds_wr + 8) = sq_q(x1);
  if (tid < 128) {
    *(uint4*)(Vs + lds_wr) = v0;
    *(uint4*)(Vs + lds_wr + 8) = v1;
    *(uint4*)(V2s + lds_wr) = sq_q(v0);
    *(uint4*)(V2s + lds_wr + 8) = sq_q(v1);
  }

  int p = 0;
  for (int k0 = 0; k0 < K; k0 += BK) {
    __syncthreads();
    const bool more = (k0 + BK) < K;
    if (more) {
      x0 = *(const uint4*)(gX + k0 + BK);
      x1 = *(const uint4*)(gX + k0 + BK + 8);
      if (tid < 128) {
        v0 = *(const uint4*)(gV + k0 + BK);
        v1 = *(const uint4*)(gV + k0 + BK + 8);
      }
      if (k0 + 2 * BK < K) __builtin_prefetch(gX + k0 + 2 * BK, 0, 3);
    }
    const bf16* xs = Xs + p * XT;
    const bf16* x2s = X2s + p * XT;
    const bf16* vs = Vs + p * VT;
    const bf16* v2s = V2s + p * VT;

    v16bf xa[2], x2a[2], vb[2], v2b[2];
#pragma unroll
    for (int i = 0; i < 2; ++i) {
      xa[i] = load_frag_a(xs + (wm + i * 16 + lm) * LDA, half);
      x2a[i] = load_frag_a(x2s + (wm + i * 16 + lm) * LDA, half);
    }
#pragma unroll
    for (int j = 0; j < 2; ++j) {
      vb[j] = load_frag_b(vs + (wn + j * 16 + lm) * LDA, half);
      v2b[j] = load_frag_b(v2s + (wn + j * 16 + lm) * LDA, half);
    }
#pragma unroll
    for (int i = 0; i < 2; ++i)
#pragma unroll
      for (int j = 0; j < 2; ++j) {
        acc1[i][j] = wmma_bf16(xa[i], vb[j], acc1[i][j]);
        acc2[i][j] = wmma_bf16(x2a[i], v2b[j], acc2[i][j]);
      }

    if (more) {
      bf16* xsn = Xs + (p ^ 1) * XT;
      bf16* x2sn = X2s + (p ^ 1) * XT;
      *(uint4*)(xsn + lds_wr) = x0;
      *(uint4*)(xsn + lds_wr + 8) = x1;
      *(uint4*)(x2sn + lds_wr) = sq_q(x0);
      *(uint4*)(x2sn + lds_wr + 8) = sq_q(x1);
      if (tid < 128) {
        bf16* vsn = Vs + (p ^ 1) * VT;
        bf16* v2sn = V2s + (p ^ 1) * VT;
        *(uint4*)(vsn + lds_wr) = v0;
        *(uint4*)(vsn + lds_wr + 8) = v1;
        *(uint4*)(v2sn + lds_wr) = sq_q(v0);
        *(uint4*)(v2sn + lds_wr + 8) = sq_q(v1);
      }
    }
    p ^= 1;
  }

#pragma unroll
  for (int i = 0; i < 2; ++i)
#pragma unroll
    for (int j = 0; j < 2; ++j) {
      const int n = wn + j * 16 + lm;
#pragma unroll
      for (int r = 0; r < 8; ++r) {
        const size_t m = m0 + wm + i * 16 + half * 8 + r;
        const float a = acc1[i][j][r], b = acc2[i][j][r];
        C[m * (size_t)ldc + ncol_off + n] = f2bf(0.5f * (a * a - b));
      }
    }
}

// ---------------- launch ----------------
extern "C" void kernel_launch(void* const* d_in, const int* in_sizes, int n_in,
                              void* d_out, int out_size, void* d_ws, size_t ws_size,
                              hipStream_t stream) {
  const float* x = (const float*)d_in[0];
  const float* Wlin = (const float*)d_in[1];
  const float* blin = (const float*)d_in[2];
  const float* V = (const float*)d_in[3];
  const float* W1 = (const float*)d_in[4];
  const float* b1 = (const float*)d_in[5];
  const float* W2 = (const float*)d_in[6];
  const float* b2 = (const float*)d_in[7];
  float* out = (float*)d_out;
  (void)in_sizes; (void)n_in; (void)out_size; (void)ws_size;

  const int M = 32 * 512;                      // 16384 tokens
  const int I = 1024, O = 512, KF = 64;
  const int CH = O + KF;                       // 576
  const int H = 2 * O;                         // 1024

  char* ws = (char*)d_ws;
  size_t off = 0;
  auto carve = [&](size_t bytes) -> char* {
    char* p = ws + off;
    off += (bytes + 255) & ~(size_t)255;
    return p;
  };
  bf16* xb    = (bf16*)carve((size_t)M * I * 2);    // x bf16        32 MB
  bf16* wlinT = (bf16*)carve((size_t)O * I * 2);    // W_lin^T
  bf16* vT    = (bf16*)carve((size_t)KF * I * 2);   // V^T
  bf16* w1T   = (bf16*)carve((size_t)H * CH * 2);   // W1^T
  bf16* w2T   = (bf16*)carve((size_t)O * H * 2);    // W2^T
  bf16* comb  = (bf16*)carve((size_t)M * CH * 2);   // combined      19 MB
  bf16* hbuf  = (bf16*)carve((size_t)M * H * 2);    // relu hidden   32 MB

  {
    long long n4 = (long long)M * I / 4;
    k_cvt_bf16<<<(unsigned)((n4 + 255) / 256), 256, 0, stream>>>(x, xb, n4);
  }
  { int n = I * O;  k_transpose_cvt<<<(n + 255) / 256, 256, 0, stream>>>(Wlin, wlinT, I, O); }
  { int n = I * KF; k_transpose_cvt<<<(n + 255) / 256, 256, 0, stream>>>(V, vT, I, KF); }
  { int n = CH * H; k_transpose_cvt<<<(n + 255) / 256, 256, 0, stream>>>(W1, w1T, CH, H); }
  { int n = H * O;  k_transpose_cvt<<<(n + 255) / 256, 256, 0, stream>>>(W2, w2T, H, O); }

  dim3 blk(256);
  // combined[:, :512] = x @ W_lin + b_lin
  gemm_wmma<false, bf16><<<dim3(M / 128, O / 128), blk, 0, stream>>>(xb, wlinT, blin, comb, CH, I);
  // combined[:, 512:576] = 0.5*((x@V)^2 - (x^2)@(V^2))
  fm_wmma<<<dim3(M / 128), blk, 0, stream>>>(xb, vT, comb, CH, O, I);
  // h = relu(combined @ W1 + b1)
  gemm_wmma<true, bf16><<<dim3(M / 128, H / 128), blk, 0, stream>>>(comb, w1T, b1, hbuf, H, CH);
  // out = h @ W2 + b2
  gemm_wmma<false, float><<<dim3(M / 128, O / 128), blk, 0, stream>>>(hbuf, w2T, b2, out, O, H);
}
